// CFG_46832323395936
// MI455X (gfx1250) — matrix operand order repaired and checked by reading, not compiled, not run on hardware
//
#include <hip/hip_runtime.h>
#include <hip/hip_bf16.h>
#include <math.h>

#define DIM        64
#define NNB        32
#define BATCH      16384
#define TILE_M     16
#define KTOT       4096          // 2 * NNB * DIM
#define A_STRIDE   144           // bf16 elems per LDS A row (128 data + 16 pad -> 288B, 16B aligned)
#define NTILES     (BATCH / TILE_M)

typedef __bf16 bf16_t;
typedef __attribute__((ext_vector_type(16))) __bf16 v16bf;
typedef __attribute__((ext_vector_type(8)))  __bf16 v8bf;
typedef __attribute__((ext_vector_type(8)))  float  v8f;

__device__ __forceinline__ v16bf cat16(v8bf lo, v8bf hi) {
  return __builtin_shufflevector(lo, hi, 0,1,2,3,4,5,6,7,8,9,10,11,12,13,14,15);
}

// B fragment (32x16 bf16): lane holds column lane%16; K = 16*part .. +16 contiguous.
__device__ __forceinline__ v16bf loadBfrag(const bf16_t* p) {
  v8bf lo = *(const v8bf*)p;
  v8bf hi = *(const v8bf*)(p + 8);
  return cat16(lo, hi);
}

// A fragment (16x32 bf16): lane holds row lane%16;
// K = 16*part .. +16 contiguous (part = lane>>4).
__device__ __forceinline__ v16bf loadAfrag(const bf16_t* rowp, int part) {
  v8bf lo = *(const v8bf*)(rowp + 16 * part);
  v8bf hi = *(const v8bf*)(rowp + 16 * part + 8);
  return cat16(lo, hi);
}

__device__ __forceinline__ float sigmoidf(float x) {
  return 1.0f / (1.0f + __expf(-x));
}

__device__ __forceinline__ float bce_term(float p, float y) {
  // torch BCELoss: log terms clamped at -100
  float lp  = fmaxf(__logf(p), -100.0f);
  float l1p = fmaxf(log1pf(-p), -100.0f);
  return -(y * lp + (1.0f - y) * l1p);
}

// ---------------- Kernel 1: gen_W f32 -> bf16 ----------------
__global__ __launch_bounds__(256) void cvtW_kernel(const float* __restrict__ W,
                                                   bf16_t* __restrict__ Wb) {
  int i = blockIdx.x * 256 + threadIdx.x;
  if (i < DIM * KTOT) Wb[i] = (bf16_t)W[i];
}

// ---------------- Kernel 2: fused gather + WMMA GEMM + loss partials ----------------
__global__ __launch_bounds__(128) void kgcn_main_kernel(
    const int*   __restrict__ users,
    const int*   __restrict__ items,
    const float* __restrict__ label,
    const int*   __restrict__ nrel,
    const int*   __restrict__ ntail,
    const float* __restrict__ user_emb,
    const float* __restrict__ entity_emb,
    const float* __restrict__ relation_emb,
    const bf16_t* __restrict__ Wb,        // [DIM][KTOT] bf16, K-contiguous rows
    const float* __restrict__ gen_b,
    float* __restrict__ partials) {

  __shared__ bf16_t Atile[2][TILE_M * A_STRIDE];   // 2 x 4.5 KB ping-pong A-chunks
  __shared__ float  u_s[TILE_M * DIM];             // 4 KB:   staged user rows
  __shared__ float  fd[TILE_M * DIM];              // 4 KB:   per-col u*false_item products
  __shared__ float  pd[TILE_M * 8];                // 512 B:  u.i partial dots
  __shared__ float  terms[TILE_M];

  const int tid  = threadIdx.x;
  const int lane = tid & 31;
  const int wave = tid >> 5;                       // 0..3 -> output column tile
  const int tile_base = blockIdx.x * TILE_M;

  const int r = tid >> 3;                          // 0..15 batch row within tile
  const int p = tid & 7;                           // 0..7  chunk within row
  const int b = tile_base + r;

  // ---- prologue: u gather, u.i partial dot ----
  {
    const int uid = users[b];
    const int iid = items[b];
    const float4* up = (const float4*)(user_emb   + (size_t)uid * DIM + p * 8);
    const float4* ip = (const float4*)(entity_emb + (size_t)iid * DIM + p * 8);
    float4 u0 = up[0], u1 = up[1];
    float4 i0 = ip[0], i1 = ip[1];
    float dp = u0.x*i0.x + u0.y*i0.y + u0.z*i0.z + u0.w*i0.w
             + u1.x*i1.x + u1.y*i1.y + u1.z*i1.z + u1.w*i1.w;
    pd[r * 8 + p] = dp;
    *(float4*)&u_s[r * DIM + p * 8]     = u0;
    *(float4*)&u_s[r * DIM + p * 8 + 4] = u1;
  }

  // loop-invariant gather addressing (branchless: p fixed per thread)
  const int*   idxp = ((p < 4) ? nrel : ntail) + (size_t)b * NNB;
  const float* tbl  = (p < 4) ? relation_emb : entity_emb;
  const int    foff = (p & 3) * 16;

  // loop-invariant fragment addressing
  const int arow  = lane & 15;                     // A row / B column-within-tile
  const int part  = lane >> 4;
  const bf16_t* wrow = Wb + (size_t)(wave * 16 + arow) * KTOT + part * 16;
  bf16_t* astore = &Atile[0][r * A_STRIDE + p * 16];
  const bf16_t* aload = &Atile[0][arow * A_STRIDE];
  const int abuf_stride = TILE_M * A_STRIDE;

  // ---- main loop: 32 neighbors, K-chunk = 128 (rel 64 | tail 64), ping-pong LDS ----
  v8f acc = {};
  for (int nb = 0; nb < NNB; ++nb) {
    const int buf = nb & 1;
    // gather 16 floats -> 16 bf16 into LDS A tile
    const int idx = idxp[nb];
    const float4* s4 = (const float4*)(tbl + (size_t)idx * DIM + foff);
    float4 f0 = s4[0], f1 = s4[1], f2 = s4[2], f3 = s4[3];
    v8bf o0, o1;
    o0[0]=(bf16_t)f0.x; o0[1]=(bf16_t)f0.y; o0[2]=(bf16_t)f0.z; o0[3]=(bf16_t)f0.w;
    o0[4]=(bf16_t)f1.x; o0[5]=(bf16_t)f1.y; o0[6]=(bf16_t)f1.z; o0[7]=(bf16_t)f1.w;
    o1[0]=(bf16_t)f2.x; o1[1]=(bf16_t)f2.y; o1[2]=(bf16_t)f2.z; o1[3]=(bf16_t)f2.w;
    o1[4]=(bf16_t)f3.x; o1[5]=(bf16_t)f3.y; o1[6]=(bf16_t)f3.z; o1[7]=(bf16_t)f3.w;
    bf16_t* dst = astore + buf * abuf_stride;
    *(v8bf*)dst       = o0;
    *(v8bf*)(dst + 8) = o1;
    __syncthreads();   // stores of buf visible; also guarantees prior reads of buf finished

    // 4 WMMA K-steps over this 128-wide chunk; B streamed from bf16 W (K-contiguous)
    const bf16_t* abase = aload + buf * abuf_stride;
    #pragma unroll
    for (int s = 0; s < 4; ++s) {
      v16bf a  = loadAfrag(abase + s * 32, part);
      v16bf bm = loadBfrag(wrow + nb * 128 + s * 32);
      acc = __builtin_amdgcn_wmma_f32_16x16x32_bf16(false, a, false, bm,
                                                    (short)0, acc, false, false);
    }
  }

  // ---- epilogue: bias + sigmoid, u * false_item products ----
  {
    const int colg = wave * 16 + arow;             // global output column 0..63
    const float bias = gen_b[colg];
    #pragma unroll
    for (int j = 0; j < 8; ++j) {
      int row = j + 8 * part;
      float fi = sigmoidf(acc[j] + bias);
      fd[row * DIM + colg] = u_s[row * DIM + colg] * fi;
    }
  }
  __syncthreads();

  // fixed-order per-row reductions (deterministic)
  if (tid < TILE_M) {
    float fdot = 0.0f;
    #pragma unroll 8
    for (int c = 0; c < DIM; ++c) fdot += fd[tid * DIM + c];
    float pdot = 0.0f;
    #pragma unroll
    for (int q = 0; q < 8; ++q) pdot += pd[tid * 8 + q];
    float y  = label[tile_base + tid];
    float p1 = sigmoidf(pdot);
    float p2 = sigmoidf(fdot);
    terms[tid] = bce_term(p1, y) + bce_term(p2, y);
  }
  __syncthreads();
  if (tid == 0) {
    float s = 0.0f;
    #pragma unroll
    for (int q = 0; q < TILE_M; ++q) s += terms[q];
    partials[blockIdx.x] = s;
  }
}

// ---------------- Kernel 3: fixed-order final reduction ----------------
__global__ void reduce_kernel(const float* __restrict__ partials, float* __restrict__ out) {
  float s = 0.0f;
  for (int i = 0; i < NTILES; ++i) s += partials[i];
  out[0] = s / (float)BATCH;
}

extern "C" void kernel_launch(void* const* d_in, const int* in_sizes, int n_in,
                              void* d_out, int out_size, void* d_ws, size_t ws_size,
                              hipStream_t stream) {
  (void)in_sizes; (void)n_in; (void)out_size; (void)ws_size;
  const int*   users        = (const int*)d_in[0];
  const int*   items        = (const int*)d_in[1];
  const float* label        = (const float*)d_in[2];
  const int*   nrel         = (const int*)d_in[3];
  const int*   ntail        = (const int*)d_in[4];
  const float* user_emb     = (const float*)d_in[5];
  const float* entity_emb   = (const float*)d_in[6];
  const float* relation_emb = (const float*)d_in[7];
  const float* gen_W        = (const float*)d_in[8];
  const float* gen_b        = (const float*)d_in[9];

  bf16_t* Wb      = (bf16_t*)d_ws;                                   // 512 KB
  float* partials = (float*)((char*)d_ws + (size_t)DIM * KTOT * 2);  // +4 KB
  float* out      = (float*)d_out;

  cvtW_kernel<<<(DIM * KTOT + 255) / 256, 256, 0, stream>>>(gen_W, Wb);
  kgcn_main_kernel<<<NTILES, 128, 0, stream>>>(users, items, label, nrel, ntail,
                                               user_emb, entity_emb, relation_emb,
                                               Wb, gen_b, partials);
  reduce_kernel<<<1, 1, 0, stream>>>(partials, out);
}
